// CausalDAG_59047210385790
// MI455X (gfx1250) — matrix operand order repaired and checked by reading, not compiled, not run on hardware
//
#include <hip/hip_runtime.h>

typedef __attribute__((ext_vector_type(16))) _Float16 v16h;
typedef __attribute__((ext_vector_type(8)))  _Float16 v8h;
typedef __attribute__((ext_vector_type(8)))  float    v8f;

union V16 { v16h v; v8h h[2]; _Float16 e[16]; };

#define WAVES 4   // 4 waves * 16 batch rows = 64 batches per workgroup

// ---------------------------------------------------------------------------
// Prep: f32 -> f16 weight conversion into workspace.
//   Ath[c][j] = A[j][c] (A transposed), padded to K=32 with zeros.
//   W1h = f16(W1) [C,G,D], W2h = f16(W2) [C,D,G]  (row-major, torch layout)
// ---------------------------------------------------------------------------
__global__ void causal_dag_prep(const float* __restrict__ A,
                                const float* __restrict__ W1,
                                const float* __restrict__ W2,
                                _Float16* __restrict__ Ath,
                                _Float16* __restrict__ W1h,
                                _Float16* __restrict__ W2h) {
    int i = blockIdx.x * blockDim.x + threadIdx.x;   // 64*256 = 16384 threads
    if (i < 512) {
        int c = i >> 5, j = i & 31;
        Ath[i] = (j < 16) ? (_Float16)A[j * 16 + c] : (_Float16)0.f;
    }
    for (int k = i; k < 16 * 32 * 64; k += 64 * 256) {
        W1h[k] = (_Float16)W1[k];
        W2h[k] = (_Float16)W2[k];
    }
}

// ---------------------------------------------------------------------------
// Fused mask -> Linear(D,G) -> ELU -> Linear(G,D) kernel.
// One wave owns 16 batch rows end-to-end; all matmuls are v_wmma_f32_16x16x32_f16.
// ---------------------------------------------------------------------------
__launch_bounds__(32 * WAVES, 1)
__global__ void causal_dag_fused(const float* __restrict__ x,     // [B,16,64]
                                 const float* __restrict__ b1,    // [16,32]
                                 const float* __restrict__ b2,    // [16,64]
                                 const _Float16* __restrict__ Ath_g,
                                 const _Float16* __restrict__ W1h_g,
                                 const _Float16* __restrict__ W2h_g,
                                 float* __restrict__ out)          // [B,16,64]
{
    __shared__ _Float16 Ath[16 * 32];            //  1 KB   A^T, K-padded
    __shared__ _Float16 W1h[16 * 32 * 64];       // 64 KB
    __shared__ _Float16 W2h[16 * 64 * 32];       // 64 KB
    __shared__ float    b1s[16 * 32];            //  2 KB
    __shared__ float    b2s[16 * 64];            //  4 KB
    __shared__ _Float16 m_h[WAVES][16 * 16 * 64];// 128 KB: m[c][b][d] per wave
    __shared__ _Float16 h_h[WAVES][16 * 32];     //  4 KB : h[b][g]   per wave

    const int tid  = threadIdx.x;
    const int lane = tid & 31;
    const int w    = tid >> 5;
    const int ln   = lane & 15;   // N-column / M-row selector within fragment
    const int lh   = lane >> 4;   // half-wave selector

    // ---- stage weights/biases (block-cooperative) ----
    for (int i = tid; i < 16 * 32 * 64 / 8; i += 32 * WAVES) {
        ((v8h*)W1h)[i] = ((const v8h*)W1h_g)[i];
        ((v8h*)W2h)[i] = ((const v8h*)W2h_g)[i];
    }
    for (int i = tid; i < 64;   i += 32 * WAVES) ((v8h*)Ath)[i] = ((const v8h*)Ath_g)[i];
    for (int i = tid; i < 512;  i += 32 * WAVES) b1s[i] = b1[i];
    for (int i = tid; i < 1024; i += 32 * WAVES) b2s[i] = b2[i];
    __syncthreads();

    const int bwave = (blockIdx.x * WAVES + w) * 16;   // first batch row of wave

    // B-fragment of the mask GEMM: A^T, K=j padded to 32 (rows 16..31 are zero).
    // B layout: lane holds column N=ln, K = {lh*8 .. lh*8+7, 16+lh*8 .. 16+lh*8+7}.
    V16 bA;
    {
        const _Float16* p = &Ath[ln * 32 + lh * 8];
        bA.h[0] = *(const v8h*)(p);
        bA.h[1] = *(const v8h*)(p + 16);
    }

    // ================= Step 1: m[c][b][d] = sum_j A[j,c] * x[b,j,d] ==========
    // Tile: M = 16 consecutive d, N = 16 concepts, K = 16 (j), zero-padded to 32.
    for (int b = 0; b < 16; ++b) {
        const size_t xbase = (size_t)(bwave + b) * (16 * 64);
        #pragma unroll
        for (int d0 = 0; d0 < 64; d0 += 16) {
            // A-frag: lane holds row d = d0+ln, K halves {lh*8 .. lh*8+7},
            //         upper K group (j>=16) = 0.
            V16 a;
            const float* xp = x + xbase + (size_t)(lh * 8) * 64 + (d0 + ln);
            #pragma unroll
            for (int t = 0; t < 8; ++t) a.e[t] = (_Float16)xp[t * 64];
            #pragma unroll
            for (int t = 8; t < 16; ++t) a.e[t] = (_Float16)0.f;

            v8f acc = {};
            acc = __builtin_amdgcn_wmma_f32_16x16x32_f16(
                    false, a.v, false, bA.v, (short)0, acc, false, false);

            // D-frag lane holds N=c=ln, M=d rows lh*8+r -> 8 contiguous d halves
            v8h s;
            #pragma unroll
            for (int r = 0; r < 8; ++r) s[r] = (_Float16)acc[r];
            *(v8h*)&m_h[w][ln * (16 * 64) + b * 64 + d0 + lh * 8] = s;
        }
    }
    __builtin_amdgcn_wave_barrier();   // m_h produced -> consumed cross-lane

    // ================= Steps 2+3 per concept ================================
    for (int c = 0; c < 16; ++c) {
        // ---- Layer 1: [16b x 32g] = m[c] (16x64) x W1[c]^T, K=64 ----
        v8f acc[2] = {v8f{}, v8f{}};
        #pragma unroll
        for (int ks = 0; ks < 2; ++ks) {
            V16 a;
            const _Float16* row = &m_h[w][c * (16 * 64) + ln * 64 + ks * 32];
            a.h[0] = *(const v8h*)(row + lh * 16);
            a.h[1] = *(const v8h*)(row + lh * 16 + 8);
            #pragma unroll
            for (int nt = 0; nt < 2; ++nt) {
                V16 bf;   // B col N=g: W1[c][g][k..k+15] contiguous
                const _Float16* wr = &W1h[c * (32 * 64) + (nt * 16 + ln) * 64 + ks * 32 + lh * 16];
                bf.h[0] = *(const v8h*)(wr);
                bf.h[1] = *(const v8h*)(wr + 8);
                acc[nt] = __builtin_amdgcn_wmma_f32_16x16x32_f16(
                            false, a.v, false, bf.v, (short)0, acc[nt], false, false);
            }
        }
        // bias + ELU, stash h as f16 [b][g]
        #pragma unroll
        for (int nt = 0; nt < 2; ++nt) {
            const float bias = b1s[c * 32 + nt * 16 + ln];
            #pragma unroll
            for (int r = 0; r < 8; ++r) {
                float v = acc[nt][r] + bias;
                v = v > 0.f ? v : expm1f(v);
                h_h[w][(lh * 8 + r) * 32 + nt * 16 + ln] = (_Float16)v;
            }
        }
        __builtin_amdgcn_wave_barrier();   // h_h cross-lane handoff

        // ---- Layer 2: [16b x 64d] = h (16x32) x W2[c]^T, K=32 ----
        V16 a3;
        {
            const _Float16* row = &h_h[w][ln * 32 + lh * 16];
            a3.h[0] = *(const v8h*)(row);
            a3.h[1] = *(const v8h*)(row + 8);
        }
        #pragma unroll
        for (int nt = 0; nt < 4; ++nt) {
            V16 bf;   // B col N=d: W2[c][d][g..g+15] contiguous
            const _Float16* wr = &W2h[c * (64 * 32) + (nt * 16 + ln) * 32 + lh * 16];
            bf.h[0] = *(const v8h*)(wr);
            bf.h[1] = *(const v8h*)(wr + 8);
            v8f acc3 = {};
            acc3 = __builtin_amdgcn_wmma_f32_16x16x32_f16(
                     false, a3.v, false, bf.v, (short)0, acc3, false, false);

            const float bias = b2s[c * 64 + nt * 16 + ln];
            float* op = out + (size_t)(bwave + lh * 8) * (16 * 64) + c * 64 + nt * 16 + ln;
            #pragma unroll
            for (int r = 0; r < 8; ++r) op[(size_t)r * (16 * 64)] = acc3[r] + bias;
        }
        __builtin_amdgcn_wave_barrier();   // before h_h reuse next c
    }
}

// ---------------------------------------------------------------------------
extern "C" void kernel_launch(void* const* d_in, const int* in_sizes, int n_in,
                              void* d_out, int out_size, void* d_ws, size_t ws_size,
                              hipStream_t stream) {
    const float* x  = (const float*)d_in[0];   // [65536,16,64]
    const float* A  = (const float*)d_in[1];   // [16,16]
    const float* W1 = (const float*)d_in[2];   // [16,32,64]
    const float* b1 = (const float*)d_in[3];   // [16,32]
    const float* W2 = (const float*)d_in[4];   // [16,64,32]
    const float* b2 = (const float*)d_in[5];   // [16,64]

    _Float16* wsh = (_Float16*)d_ws;
    _Float16* Ath = wsh;                        // 512 halves
    _Float16* W1h = wsh + 512;                  // 32768 halves
    _Float16* W2h = wsh + 512 + 32768;          // 32768 halves

    causal_dag_prep<<<64, 256, 0, stream>>>(A, W1, W2, Ath, W1h, W2h);

    // 65536 batches / 64 per workgroup = 1024 workgroups, 128 threads (4 waves)
    causal_dag_fused<<<1024, 32 * WAVES, 0, stream>>>(
        x, b1, b2, Ath, W1h, W2h, (float*)d_out);
}
